// MyCustomModel_73641509257424
// MI455X (gfx1250) — compile-verified, matrix operand order checked by reading, not run on hardware
//
#include <hip/hip_runtime.h>

// ---------------------------------------------------------------------------
// MI455X (gfx1250) implementation.
//
// Piece 1: conv2d(NCHW,VALID,3x3)+ReLU as implicit GEMM:
//   - weights staged per K-tile by the Tensor Data Mover (tensor_load_to_lds,
//     TENSORcnt) as an affine 64x64 fp32 tile, overlapped with the per-lane
//     im2col staging of x,
//   - both operands split fp32 -> bf16 hi/lo in LDS,
//   - 3x v_wmma_f32_16x16x32_bf16 per 16x16x32 step (hi*hi + hi*lo + lo*hi)
//     ~ fp32 accuracy at 8/3 the native f32-WMMA (16x16x4) rate.
// Piece 2: concat(a,b) as a float4 streaming copy.
// ---------------------------------------------------------------------------

typedef __attribute__((ext_vector_type(16))) __bf16 v16bf;
typedef __attribute__((ext_vector_type(8)))  float  v8f;
typedef __attribute__((ext_vector_type(4)))  unsigned int u32x4;
typedef __attribute__((ext_vector_type(8)))  int  i32x8;
typedef __attribute__((ext_vector_type(4)))  int  i32x4;

#define W_OUT    126
#define HW_OUT   15876            // 126*126
#define TOTAL_P  508032           // 32 * 126*126
#define C_IN     64
#define OC       128
#define K_TOT    576              // 64 * 9
#define BM       64
#define BN       64
#define BK       64
#define LDS_STR  66               // 33 dwords: odd stride -> no bank conflicts

#define CONV_OUT_ELEMS 65028096   // 32*128*15876

union Frag {
  v16bf        v;
  unsigned int u[8];
};

__device__ __forceinline__ unsigned short f2bf(float f) {
  unsigned int u = __float_as_uint(f);
  u += 0x7FFFu + ((u >> 16) & 1u);          // round-to-nearest-even
  return (unsigned short)(u >> 16);
}
__device__ __forceinline__ float bf2f(unsigned short h) {
  return __uint_as_float(((unsigned int)h) << 16);
}

// Build a 16x32 bf16 A/B fragment from an LDS row ([row][K], K contiguous).
// ISA layout (CDNA5 ISA): VGPR v<4 -> K=2v,2v+1 ; v>=4 -> K=16+2(v-4);
// lanes 16..31 add +8 to K.  All pairs are even -> one dword read per VGPR.
__device__ __forceinline__ v16bf frag_from_lds(const unsigned short* base, int hi8) {
  Frag f;
#pragma unroll
  for (int v = 0; v < 8; ++v) {
    const int kp = ((v < 4) ? (2 * v) : (16 + 2 * (v - 4))) + hi8;
    f.u[v] = *(const unsigned int*)(base + kp);
  }
  return f.v;
}

// Tensor Data Mover: DMA one 64(k) x 64(oc) fp32 weight tile into LDS.
// D# packing per CDNA5 ISA (2D tensor, groups 2/3 zero):
//   tensor = w viewed as 128 rows x 576 cols fp32, row stride 576;
//   tile   = 64 cols (k, contiguous) x 64 rows (oc); LDS gets Wraw[oc][k].
__device__ __forceinline__ void tdm_load_w_tile(const float* w, int Mblock, int kb,
                                                unsigned lds_byte_addr) {
  const unsigned long long ga =
      (unsigned long long)(size_t)w +
      (unsigned long long)((unsigned)(Mblock * K_TOT + kb)) * 4ull;
  u32x4 g0;
  g0[0] = 1u;                                   // count=1, user descriptor
  g0[1] = lds_byte_addr;                        // bits [63:32]  lds_addr
  g0[2] = (unsigned)(ga & 0xFFFFFFFFull);       // bits [95:64]  global_addr lo
  g0[3] = (unsigned)((ga >> 32) & 0x1FFFFFFull) // bits [120:96] global_addr hi
        | (2u << 30);                           // bits [127:126] type = 2
  i32x8 g1;
  g1[0] = (int)(2u << 16);                      // data_size = 4B; wg_mask = 0
  g1[1] = (int)((576u & 0xFFFFu) << 16);        // tensor_dim0[15:0] = 576
  g1[2] = (int)(((576u >> 16) & 0xFFFFu) | ((128u & 0xFFFFu) << 16)); // dim0 hi | dim1 lo
  g1[3] = (int)(((128u >> 16) & 0xFFFFu) | (64u << 16));              // dim1 hi | tile_dim0=64
  g1[4] = (int)(64u);                           // tile_dim1 = 64, tile_dim2 = 0
  g1[5] = (int)576;                             // tensor_dim0_stride[31:0]
  g1[6] = 0;                                    // stride0 hi | dim1_stride lo
  g1[7] = 0;
  const i32x4 z4 = {0, 0, 0, 0};
#if defined(__clang_major__) && (__clang_major__ >= 23)
  const i32x8 z8 = {0, 0, 0, 0, 0, 0, 0, 0};
  __builtin_amdgcn_tensor_load_to_lds(g0, g1, z4, z4, z8, 0);
#else
  __builtin_amdgcn_tensor_load_to_lds(g0, g1, z4, z4, 0);
#endif
}

__global__ __launch_bounds__(128) void conv3x3_wmma_bf16x3(
    const float* __restrict__ x,   // (32,64,128,128)
    const float* __restrict__ w,   // (128,64,3,3) -> (128,576) K-contiguous
    float* __restrict__ out) {     // (32,128,15876)
  __shared__ float          Wraw[BM * BK];      // raw fp32 weight tile (TDM dest)
  __shared__ unsigned short Ah[BM * LDS_STR];   // weights, hi
  __shared__ unsigned short Al[BM * LDS_STR];   // weights, lo
  __shared__ unsigned short Bh[BN * LDS_STR];   // im2col x, hi  ([p][k])
  __shared__ unsigned short Bl[BN * LDS_STR];   // im2col x, lo

  const int tid  = threadIdx.x;
  const int lane = tid & 31;
  const int wave = tid >> 5;          // 4 waves
  const int wm   = wave >> 1;         // 0..1 : 32-row strip of the 64x64 tile
  const int wn   = wave & 1;          // 0..1 : 32-col strip

  const int Mblock = blockIdx.y * BM;           // out-channel tile
  const int Nblock = blockIdx.x * BN;           // spatial tile

  const unsigned wraw_lds = (unsigned)(size_t)&Wraw[0];   // LDS byte address

  v8f acc[2][2];
#pragma unroll
  for (int i = 0; i < 2; ++i)
#pragma unroll
    for (int j = 0; j < 2; ++j)
      acc[i][j] = {};

  // Per-thread fixed coordinates for the stagers.
  const int klA = tid & 63;           // A: K fastest
  const int ml0 = tid >> 6;
  const int pl  = tid & 63;           // B: P fastest (coalesced over x cols)
  const int p   = Nblock + pl;
  const bool pvalid = (p < TOTAL_P);
  int n = 0, h0 = 0, w0 = 0;
  if (pvalid) {
    n = p / HW_OUT;
    const int hw = p - n * HW_OUT;
    h0 = hw / W_OUT;
    w0 = hw - h0 * W_OUT;
  }

  const int hi8 = (lane >> 4) << 3;   // +8 K-offset for lanes 16..31
  const int m16 = lane & 15;

  for (int kb = 0; kb < K_TOT; kb += BK) {
    // ---- kick off the weight-tile DMA (one wave issues; TENSORcnt tracks) ----
    if (wave == 0) tdm_load_w_tile(w, Mblock, kb, wraw_lds);

    // ---- overlapped: stage 64(K) x 64(P) im2col tile, [p][k], bf16 hi/lo ----
#pragma unroll 4
    for (int i = 0; i < 32; ++i) {
      const int kl = (tid >> 6) + 2 * i;
      const int k  = kb + kl;
      const int c  = k / 9;
      const int rs = k - c * 9;
      const int r  = rs / 3;
      const int s  = rs - r * 3;
      float v = 0.f;
      if (pvalid)
        v = x[((n * C_IN + c) * 128 + (h0 + r)) * 128 + (w0 + s)];
      const unsigned short h = f2bf(v);
      Bh[pl * LDS_STR + kl] = h;
      Bl[pl * LDS_STR + kl] = f2bf(v - bf2f(h));
    }

    if (wave == 0) __builtin_amdgcn_s_wait_tensorcnt(0);
    __syncthreads();                  // Wraw + B tiles visible to all waves

    // ---- split the raw weight tile into bf16 hi/lo ----
#pragma unroll 4
    for (int i = 0; i < 32; ++i) {
      const int ml = ml0 + 2 * i;
      const float v = Wraw[ml * BK + klA];
      const unsigned short h = f2bf(v);
      Ah[ml * LDS_STR + klA] = h;
      Al[ml * LDS_STR + klA] = f2bf(v - bf2f(h));
    }
    __syncthreads();                  // Ah/Al ready

    // ---- compute: 2 K-steps of 32, 2x2 16x16 tiles, 3 WMMAs each ----
#pragma unroll
    for (int kk = 0; kk < BK; kk += 32) {
      v16bf ah[2], al_[2], bh[2], bl_[2];
#pragma unroll
      for (int i = 0; i < 2; ++i) {
        const int ar = (wm * 32 + i * 16 + m16) * LDS_STR + kk;
        ah[i]  = frag_from_lds(&Ah[ar], hi8);
        al_[i] = frag_from_lds(&Al[ar], hi8);
        const int br = (wn * 32 + i * 16 + m16) * LDS_STR + kk;
        bh[i]  = frag_from_lds(&Bh[br], hi8);
        bl_[i] = frag_from_lds(&Bl[br], hi8);
      }
#pragma unroll
      for (int i = 0; i < 2; ++i)
#pragma unroll
        for (int j = 0; j < 2; ++j) {
          acc[i][j] = __builtin_amdgcn_wmma_f32_16x16x32_bf16(
              false, ah[i],  false, bh[j],  (short)0, acc[i][j], false, false);
          acc[i][j] = __builtin_amdgcn_wmma_f32_16x16x32_bf16(
              false, ah[i],  false, bl_[j], (short)0, acc[i][j], false, false);
          acc[i][j] = __builtin_amdgcn_wmma_f32_16x16x32_bf16(
              false, al_[i], false, bh[j],  (short)0, acc[i][j], false, false);
        }
    }
    __syncthreads();                  // protect all tiles before next overwrite
  }

  // ---- fused ReLU + store (C/D layout: VGPR r -> M = r + 8*(lane>=16)) ----
  const int rowAdd = hi8;             // 0 or 8
#pragma unroll
  for (int j = 0; j < 2; ++j) {
    const int pj = Nblock + wn * 32 + j * 16 + m16;
    if (pj < TOTAL_P) {
      const int nj  = pj / HW_OUT;
      const int hwj = pj - nj * HW_OUT;
      const int obase = nj * (OC * HW_OUT) + hwj;
#pragma unroll
      for (int i = 0; i < 2; ++i) {
        const int ocb = Mblock + wm * 32 + i * 16 + rowAdd;
#pragma unroll
        for (int r = 0; r < 8; ++r) {
          const float v = acc[i][j][r];
          out[obase + (ocb + r) * HW_OUT] = v > 0.f ? v : 0.f;
        }
      }
    }
  }
}

// ---- concat(a, b) along last dim: pure streaming copy in float4 ----
__global__ __launch_bounds__(256) void concat_rows_f4(
    const float4* __restrict__ a, const float4* __restrict__ b,
    float4* __restrict__ out) {
  const int t = blockIdx.x * blockDim.x + threadIdx.x;  // 0 .. 1024*512-1
  const int r = t >> 9;            // row
  const int q = t & 511;           // float4 col within 2048-float row
  const float4 v = (q < 256) ? a[r * 256 + q] : b[r * 256 + (q - 256)];
  out[r * 512 + q] = v;
}

extern "C" void kernel_launch(void* const* d_in, const int* in_sizes, int n_in,
                              void* d_out, int out_size, void* d_ws, size_t ws_size,
                              hipStream_t stream) {
  const float* x = (const float*)d_in[0];
  const float* w = (const float*)d_in[1];
  const float* a = (const float*)d_in[2];
  const float* b = (const float*)d_in[3];
  // d_in[4] = 'shape' (int64) is static in the reference; not needed.

  float* out = (float*)d_out;

  // Conv: M tiles = 128/64 = 2, N tiles = ceil(508032/64) = 7938.
  dim3 grid((TOTAL_P + BN - 1) / BN, OC / BM);
  conv3x3_wmma_bf16x3<<<grid, 128, 0, stream>>>(x, w, out);

  // Concat into the second output region.
  float* out2 = out + CONV_OUT_ELEMS;
  const int total_f4 = 1024 * 512;   // 1024 rows * 512 float4 per row
  concat_rows_f4<<<total_f4 / 256, 256, 0, stream>>>(
      (const float4*)a, (const float4*)b, (float4*)out2);
}
